// PointNetFeaturePropagation_15788299780824
// MI455X (gfx1250) — matrix operand order repaired and checked by reading, not compiled, not run on hardware
//
#include <hip/hip_runtime.h>
#include <hip/hip_bf16.h>

typedef __attribute__((ext_vector_type(16))) _Float16 v16h;
typedef __attribute__((ext_vector_type(8)))  float    v8f;
typedef __attribute__((ext_vector_type(2)))  _Float16 h2;
typedef unsigned int u32;

#define Bv 16
#define Nv 4096
#define Sv 1024
#define Dv 256
#define O1v 256
#define O2v 128

// ---------------------------------------------------------------------------
// Fragment helpers: build v16h WMMA operands from uint pairs (ISA 7.12.2
// layouts). hs = lane>>4 (half-wave select).
// A-matrix 16x32 f16: VGPR j<4 -> K = 2j + 8*hs ; j>=4 -> K = 16 + 2(j-4) + 8*hs
// B-matrix 32x16 f16: VGPR j<4 -> K = 2j + 8*hs ; j>=4 -> K = 16 + 2(j-4) + 8*hs (same as A)
// ---------------------------------------------------------------------------
union FragU { u32 u[8]; v16h h; };

__device__ __forceinline__ v16h load_a_frag(const u32* W, int m, int kbu, int hs) {
    const u32* p = W + m * 128 + kbu + 4 * hs;   // kbu = kb>>1 (u32 units)
    FragU f;
    f.u[0] = p[0];  f.u[1] = p[1];  f.u[2] = p[2];  f.u[3] = p[3];
    f.u[4] = p[8];  f.u[5] = p[9];  f.u[6] = p[10]; f.u[7] = p[11];
    return f.h;
}

__device__ __forceinline__ v16h load_b_frag(const u32* xs, int nloc, int kbu, int hs) {
    const u32* p = xs + nloc * 132 + kbu + 4 * hs;  // 132 u32 = 264 f16 padded row
    FragU f;
#pragma unroll
    for (int j = 0; j < 8; ++j) f.u[j] = p[j + (j & 4)];
    return f.h;
}

// ---------------------------------------------------------------------------
// Tiled transpose: in[b][r][c] (f32) -> out[b][c][r] (f16).  R,C multiples of 32.
// ---------------------------------------------------------------------------
__global__ __launch_bounds__(256) void transpose_to_h(const float* __restrict__ in,
                                                      _Float16* __restrict__ out,
                                                      int R, int C) {
    __shared__ float t[32][33];
    int b  = blockIdx.z;
    int c0 = blockIdx.x * 32, r0 = blockIdx.y * 32;
    const float* ib = in + (size_t)b * R * C;
#pragma unroll
    for (int k = 0; k < 32; k += 8)
        t[threadIdx.y + k][threadIdx.x] =
            ib[(size_t)(r0 + threadIdx.y + k) * C + c0 + threadIdx.x];
    __syncthreads();
    _Float16* ob = out + (size_t)b * R * C;
#pragma unroll
    for (int k = 0; k < 32; k += 8)
        ob[(size_t)(c0 + threadIdx.y + k) * R + r0 + threadIdx.x] =
            (_Float16)t[threadIdx.x][threadIdx.y + k];
}

__global__ void cvt_h(const float* __restrict__ in, _Float16* __restrict__ out, int n) {
    int i = blockIdx.x * 256 + threadIdx.x;
    if (i < n) out[i] = (_Float16)in[i];
}

// ---------------------------------------------------------------------------
// Interp: one wave per (b,n). 3-NN over S=1024 via per-lane sorted top-3 +
// butterfly shfl_xor merge, then coalesced weighted gather (p2t is n-major).
// X out: [b][n][c] f16.
// ---------------------------------------------------------------------------
__global__ __launch_bounds__(256) void interp3(const float* __restrict__ xyz1,
                                               const float* __restrict__ xyz2,
                                               const _Float16* __restrict__ p2t,
                                               const _Float16* __restrict__ p1t,
                                               _Float16* __restrict__ xout) {
    int wave = (int)((blockIdx.x * 256u + threadIdx.x) >> 5);
    int lane = threadIdx.x & 31;
    int b = wave >> 12;          // / Nv
    int n = wave & (Nv - 1);

    float px = xyz1[(size_t)b * 3 * Nv + n];
    float py = xyz1[(size_t)b * 3 * Nv + Nv + n];
    float pz = xyz1[(size_t)b * 3 * Nv + 2 * Nv + n];
    const float* bx = xyz2 + (size_t)b * 3 * Sv;

    float d0 = 3.0e38f, d1 = 3.0e38f, d2 = 3.0e38f;
    int   i0 = 0, i1 = 0, i2 = 0;
#pragma unroll 4
    for (int t = 0; t < 32; ++t) {
        int s = t * 32 + lane;
        float dx = px - bx[s];
        float dy = py - bx[Sv + s];
        float dz = pz - bx[2 * Sv + s];
        float dd = dx * dx + dy * dy + dz * dz;
        if (dd < d2) {
            if (dd < d1) {
                d2 = d1; i2 = i1;
                if (dd < d0) { d1 = d0; i1 = i0; d0 = dd; i0 = s; }
                else         { d1 = dd; i1 = s; }
            } else { d2 = dd; i2 = s; }
        }
    }
    // butterfly merge of sorted triples across the wave
    for (int off = 16; off > 0; off >>= 1) {
        float e0 = __shfl_xor(d0, off, 32);
        float e1 = __shfl_xor(d1, off, 32);
        float e2 = __shfl_xor(d2, off, 32);
        int   j0 = __shfl_xor(i0, off, 32);
        int   j1 = __shfl_xor(i1, off, 32);
        int   j2 = __shfl_xor(i2, off, 32);
        float m0, m1, m2; int q0, q1, q2;
        if (d0 <= e0) {
            m0 = d0; q0 = i0;
            if (d1 <= e0) {
                m1 = d1; q1 = i1;
                if (d2 <= e0) { m2 = d2; q2 = i2; } else { m2 = e0; q2 = j0; }
            } else {
                m1 = e0; q1 = j0;
                if (d1 <= e1) { m2 = d1; q2 = i1; } else { m2 = e1; q2 = j1; }
            }
        } else {
            m0 = e0; q0 = j0;
            if (e1 <= d0) {
                m1 = e1; q1 = j1;
                if (e2 <= d0) { m2 = e2; q2 = j2; } else { m2 = d0; q2 = i0; }
            } else {
                m1 = d0; q1 = i0;
                if (d1 <= e1) { m2 = d1; q2 = i1; } else { m2 = e1; q2 = j1; }
            }
        }
        d0 = m0; d1 = m1; d2 = m2; i0 = q0; i1 = q1; i2 = q2;
    }
    float r0 = 1.0f / (d0 + 1e-8f);
    float r1 = 1.0f / (d1 + 1e-8f);
    float r2 = 1.0f / (d2 + 1e-8f);
    float rs = 1.0f / (r0 + r1 + r2);
    float w0 = r0 * rs, w1 = r1 * rs, w2 = r2 * rs;

    const _Float16* g0 = p2t + ((size_t)(b * Sv + i0)) * Dv;
    const _Float16* g1 = p2t + ((size_t)(b * Sv + i1)) * Dv;
    const _Float16* g2 = p2t + ((size_t)(b * Sv + i2)) * Dv;
    const _Float16* p1 = p1t + ((size_t)(b * Nv + n)) * Dv;
    _Float16*       xo = xout + ((size_t)(b * Nv + n)) * Dv;
#pragma unroll
    for (int j = 0; j < 8; ++j) {
        int c = j * 32 + lane;
        float v = w0 * (float)g0[c] + w1 * (float)g1[c] + w2 * (float)g2[c] + (float)p1[c];
        xo[c] = (_Float16)v;
    }
}

// ---------------------------------------------------------------------------
// GEMM1: Y1[b][n][o] (f16) = relu( BN( W1[256x256] * X[b][:, n] ) )
// Workgroup: 512 threads = 16 waves, covers 64 columns of n. K = 256.
// ---------------------------------------------------------------------------
__global__ __launch_bounds__(512) void gemm_bn_relu_h(const _Float16* __restrict__ X,
                                                      const u32* __restrict__ Wq,
                                                      const float* __restrict__ bb,
                                                      const float* __restrict__ gg,
                                                      const float* __restrict__ be,
                                                      const float* __restrict__ mm,
                                                      const float* __restrict__ vv,
                                                      _Float16* __restrict__ Y) {
    __shared__ u32 xs[64 * 132];
    int b  = blockIdx.x >> 6;
    int n0 = (blockIdx.x & 63) << 6;
    const u32* gx = (const u32*)X + ((size_t)(b * Nv + n0)) * 128;
    __builtin_prefetch(Wq, 0, 0);
    for (int i = threadIdx.x; i < 64 * 128; i += 512) {
        int row = i >> 7, col = i & 127;
        xs[row * 132 + col] = gx[row * 128 + col];
    }
    __syncthreads();

    int wave = threadIdx.x >> 5, lane = threadIdx.x & 31;
    int hs = lane >> 4, l15 = lane & 15;
    int mtile = wave;                     // 0..15 -> O1 = 256

    v8f acc[4] = {};
#pragma unroll
    for (int kb = 0; kb < 256; kb += 32) {
        int kbu = kb >> 1;
        v16h a = load_a_frag(Wq, mtile * 16 + l15, kbu, hs);
#pragma unroll
        for (int nt = 0; nt < 4; ++nt) {
            v16h bf = load_b_frag(xs, nt * 16 + l15, kbu, hs);
            acc[nt] = __builtin_amdgcn_wmma_f32_16x16x32_f16(
                false, a, false, bf, (short)0, acc[nt], false, false);
        }
    }

    float sc[8], bi[8];
#pragma unroll
    for (int r = 0; r < 8; ++r) {
        int o = mtile * 16 + r + 8 * hs;
        float s = gg[o] * rsqrtf(vv[o] + 1e-5f);
        sc[r] = s;
        bi[r] = (bb[o] - mm[o]) * s + be[o];
    }
#pragma unroll
    for (int nt = 0; nt < 4; ++nt) {
        int n = n0 + nt * 16 + l15;
        size_t base = ((size_t)(b * Nv + n)) * O1v + mtile * 16 + 8 * hs;
#pragma unroll
        for (int r = 0; r < 8; r += 2) {
            float f0 = fmaxf(acc[nt][r] * sc[r] + bi[r], 0.0f);
            float f1 = fmaxf(acc[nt][r + 1] * sc[r + 1] + bi[r + 1], 0.0f);
            h2 pk = { (_Float16)f0, (_Float16)f1 };
            *(h2*)(Y + base + r) = pk;
        }
    }
}

// ---------------------------------------------------------------------------
// GEMM2: out[b][o][n] (f32) = relu( BN( W2[128x256] * Y1[b][:, n] ) )
// Workgroup: 256 threads = 8 waves, covers 64 columns of n. K = 256.
// ---------------------------------------------------------------------------
__global__ __launch_bounds__(256) void gemm_bn_relu_f(const _Float16* __restrict__ X,
                                                      const u32* __restrict__ Wq,
                                                      const float* __restrict__ bb,
                                                      const float* __restrict__ gg,
                                                      const float* __restrict__ be,
                                                      const float* __restrict__ mm,
                                                      const float* __restrict__ vv,
                                                      float* __restrict__ out) {
    __shared__ u32 xs[64 * 132];
    int b  = blockIdx.x >> 6;
    int n0 = (blockIdx.x & 63) << 6;
    const u32* gx = (const u32*)X + ((size_t)(b * Nv + n0)) * 128;
    __builtin_prefetch(Wq, 0, 0);
    for (int i = threadIdx.x; i < 64 * 128; i += 256) {
        int row = i >> 7, col = i & 127;
        xs[row * 132 + col] = gx[row * 128 + col];
    }
    __syncthreads();

    int wave = threadIdx.x >> 5, lane = threadIdx.x & 31;
    int hs = lane >> 4, l15 = lane & 15;
    int mtile = wave;                     // 0..7 -> O2 = 128

    v8f acc[4] = {};
#pragma unroll
    for (int kb = 0; kb < 256; kb += 32) {
        int kbu = kb >> 1;
        v16h a = load_a_frag(Wq, mtile * 16 + l15, kbu, hs);
#pragma unroll
        for (int nt = 0; nt < 4; ++nt) {
            v16h bf = load_b_frag(xs, nt * 16 + l15, kbu, hs);
            acc[nt] = __builtin_amdgcn_wmma_f32_16x16x32_f16(
                false, a, false, bf, (short)0, acc[nt], false, false);
        }
    }

    float sc[8], bi[8];
#pragma unroll
    for (int r = 0; r < 8; ++r) {
        int o = mtile * 16 + r + 8 * hs;
        float s = gg[o] * rsqrtf(vv[o] + 1e-5f);
        sc[r] = s;
        bi[r] = (bb[o] - mm[o]) * s + be[o];
    }
#pragma unroll
    for (int nt = 0; nt < 4; ++nt) {
        int n = n0 + nt * 16 + l15;
#pragma unroll
        for (int r = 0; r < 8; ++r) {
            int o = mtile * 16 + r + 8 * hs;
            out[(size_t)b * O2v * Nv + (size_t)o * Nv + n] =
                fmaxf(acc[nt][r] * sc[r] + bi[r], 0.0f);
        }
    }
}

// ---------------------------------------------------------------------------
extern "C" void kernel_launch(void* const* d_in, const int* in_sizes, int n_in,
                              void* d_out, int out_size, void* d_ws, size_t ws_size,
                              hipStream_t stream) {
    (void)in_sizes; (void)n_in; (void)out_size; (void)ws_size;
    const float* xyz1    = (const float*)d_in[0];
    const float* xyz2    = (const float*)d_in[1];
    const float* points1 = (const float*)d_in[2];
    const float* points2 = (const float*)d_in[3];
    const float* W1 = (const float*)d_in[4];
    const float* b1 = (const float*)d_in[5];
    const float* g1 = (const float*)d_in[6];
    const float* be1 = (const float*)d_in[7];
    const float* m1 = (const float*)d_in[8];
    const float* v1 = (const float*)d_in[9];
    const float* W2 = (const float*)d_in[10];
    const float* b2 = (const float*)d_in[11];
    const float* g2 = (const float*)d_in[12];
    const float* be2 = (const float*)d_in[13];
    const float* m2 = (const float*)d_in[14];
    const float* v2 = (const float*)d_in[15];

    char* ws = (char*)d_ws;
    const size_t szP2T = (size_t)Bv * Sv * Dv * 2;   //  8 MB
    const size_t szP1T = (size_t)Bv * Nv * Dv * 2;   // 32 MB
    const size_t szX   = szP1T;                      // 32 MB
    const size_t szY1  = (size_t)Bv * Nv * O1v * 2;  // 32 MB
    _Float16* p2t = (_Float16*)ws;
    _Float16* p1t = (_Float16*)(ws + szP2T);
    _Float16* Xh  = (_Float16*)(ws + szP2T + szP1T);
    _Float16* Y1h = (_Float16*)(ws + szP2T + szP1T + szX);
    _Float16* W1h = (_Float16*)(ws + szP2T + szP1T + szX + szY1);
    _Float16* W2h = (_Float16*)(ws + szP2T + szP1T + szX + szY1 + (size_t)O1v * Dv * 2);

    // 1. transpose points2 (B,D,S)->(B,S,D) f16 ; points1 (B,D,N)->(B,N,D) f16
    transpose_to_h<<<dim3(Sv / 32, Dv / 32, Bv), dim3(32, 8), 0, stream>>>(points2, p2t, Dv, Sv);
    transpose_to_h<<<dim3(Nv / 32, Dv / 32, Bv), dim3(32, 8), 0, stream>>>(points1, p1t, Dv, Nv);

    // 2. weights -> f16 row-major
    cvt_h<<<(O1v * Dv + 255) / 256, 256, 0, stream>>>(W1, W1h, O1v * Dv);
    cvt_h<<<(O2v * O1v + 255) / 256, 256, 0, stream>>>(W2, W2h, O2v * O1v);

    // 3. 3-NN interpolation + points1 add -> X[b][n][c] f16
    interp3<<<(Bv * Nv) / 8, 256, 0, stream>>>(xyz1, xyz2, p2t, p1t, Xh);

    // 4. conv1 + BN + ReLU (WMMA) -> Y1[b][n][o] f16
    gemm_bn_relu_h<<<Bv * (Nv / 64), 512, 0, stream>>>(Xh, (const u32*)W1h,
                                                       b1, g1, be1, m1, v1, Y1h);

    // 5. conv2 + BN + ReLU (WMMA) -> out[b][o][n] f32
    gemm_bn_relu_f<<<Bv * (Nv / 64), 256, 0, stream>>>(Y1h, (const u32*)W2h,
                                                       b2, g2, be2, m2, v2, (float*)d_out);
}
